// MonarchAttention_154618823094
// MI455X (gfx1250) — matrix operand order repaired and checked by reading, not compiled, not run on hardware
//
#include <hip/hip_runtime.h>

// ---------------------------------------------------------------------------
// Monarch causal linear attention for MI455X (gfx1250, wave32, WMMA bf16).
//
// Pipeline:
//   0) fp32 -> bf16 converts (X, 6 weights)
//   1) WMMA GEMM (K=64 LDS stages, 8 wmma/stage/wave, async-to-LDS A tiles):
//      q/k1/v1/k2/v2 projections (SiLU fused), bf16 [H][L][64] head layout
//   2) causal linear attention stage 1 (normalizer = rowsum of tril(QK^T),
//      i.e. q . cumsum(k) without a serial scan)
//   3) Monarch head shuffle  y1[h1][l][h2*4+d] -> q2[h2][l][h1*4+d]
//   4) causal linear attention stage 2, output bf16 [L][DIM]
//   5) WMMA GEMM: y2 @ Wo -> fp32 d_out
// ---------------------------------------------------------------------------

typedef __bf16 bf16;
typedef __attribute__((ext_vector_type(16))) __bf16 v16bf;
typedef __attribute__((ext_vector_type(8)))  __bf16 v8bf;
typedef __attribute__((ext_vector_type(8)))  float  v8f;
typedef int ma_v4i __attribute__((vector_size(16)));   // int4, per builtin sig

#define SEQ   2048
#define EMB   1024
#define NHEAD 16
#define HDIM  64

// ---- CDNA5 async copy-to-LDS (GLOBAL_LOAD_ASYNC_TO_LDS_B128, ASYNCcnt) ----
// Probe: builtin takes (AS1 int4*, AS3 int4*, imm offset, imm cpol).
#if __has_builtin(__builtin_amdgcn_global_load_async_to_lds_b128) && \
    __has_builtin(__builtin_amdgcn_s_wait_asynccnt)
#define MA_ASYNC 1
#endif

static __device__ __forceinline__ void copy16_to_lds(const bf16* g, bf16* l) {
#ifdef MA_ASYNC
  __builtin_amdgcn_global_load_async_to_lds_b128(
      (__attribute__((address_space(1))) ma_v4i*)g,
      (__attribute__((address_space(3))) ma_v4i*)l, 0, 0);
#else
  *(v8bf*)l = *(const v8bf*)g;
#endif
}

static __device__ __forceinline__ void async_fence() {
#ifdef MA_ASYNC
  __builtin_amdgcn_s_wait_asynccnt(0);   // async LDS writes visible pre-barrier
#endif
}

// ----- WMMA fragment gathers from LDS (layouts per CDNA5 ISA) ----
// A (16x32 bf16): M = lane&15; lanes 0-15: K in {kb..kb+7, kb+16..kb+23},
// kb=0; lanes 16-31: kb=8. LDS row-major [row][k], k contiguous.
static __device__ __forceinline__ v16bf frag_a(const bf16* lds, int ld,
                                               int row_base, int koff, int lane) {
  const int row = row_base + (lane & 15);
  const int kb  = koff + ((lane & 16) ? 8 : 0);
  const bf16* p = lds + row * ld + kb;
  v16bf a;
#pragma unroll
  for (int e = 0; e < 8; ++e)  a[e] = p[e];        // K = kb + e
#pragma unroll
  for (int e = 8; e < 16; ++e) a[e] = p[e + 8];    // K = kb + 16 + (e-8)
  return a;
}

// B (32x16 bf16): N = lane&15; lanes 0-15: K in {kb..kb+7, kb+16..kb+23},
// kb=0; lanes 16-31: kb=8. LDS staged [n][k], k contiguous.
static __device__ __forceinline__ v16bf frag_b(const bf16* lds, int ld,
                                               int col_base, int koff, int lane) {
  const int col = col_base + (lane & 15);
  const int kb  = koff + ((lane & 16) ? 8 : 0);
  const bf16* p = lds + col * ld + kb;
  v16bf b;
#pragma unroll
  for (int e = 0; e < 8; ++e)  b[e] = p[e];
#pragma unroll
  for (int e = 8; e < 16; ++e) b[e] = p[e + 8];
  return b;
}

#define WMMA_BF16(A, B, C) \
  __builtin_amdgcn_wmma_f32_16x16x32_bf16(false, (A), false, (B), (short)0, (C), false, false)

// ---------------------------------------------------------------------------
__global__ void ma_cvt_bf16(const float* __restrict__ in, bf16* __restrict__ out, int n) {
  int i = blockIdx.x * 256 + threadIdx.x;
  if (i < n) out[i] = (bf16)in[i];
}

// ---------------------------------------------------------------------------
// WMMA GEMM:  C[M,N] = act(A[M,K] @ W[K,N])
//   block tile 128x64, 256 threads = 8 waves (4x2 of 32x32 wave tiles),
//   K staged 64 at a time -> 8 wmma per wave between barriers.
// ---------------------------------------------------------------------------
__global__ __launch_bounds__(256) void ma_gemm_bf16(
    const bf16* __restrict__ A, const bf16* __restrict__ W,
    int M, int N, int K, int do_silu,
    bf16* __restrict__ out_head, float* __restrict__ out_f32) {
  __shared__ bf16 sA[128 * 64];   // [m][k]   16 KB
  __shared__ bf16 sB[64 * 64];    // [n][k]    8 KB (transposed stage)

  const int t    = threadIdx.x;
  const int lane = t & 31;
  const int wave = t >> 5;
  const int wm   = wave >> 1;           // 0..3
  const int wn   = wave & 1;            // 0..1
  const int bm   = blockIdx.y * 128;
  const int bn   = blockIdx.x * 64;

  v8f acc[2][2] = {};

  for (int k0 = 0; k0 < K; k0 += 64) {
    // A tile 128x64: 1024 x 16B chunks, async DMA straight into LDS
#pragma unroll
    for (int i = 0; i < 4; ++i) {
      int c   = t + i * 256;
      int row = c >> 3;                 // 8 chunks per 64-elem row
      int off = (c & 7) * 8;
      copy16_to_lds(A + (size_t)(bm + row) * K + k0 + off, sA + row * 64 + off);
    }
    // B tile 64x64 -> LDS [n][k] (transpose: staged through VGPRs)
#pragma unroll
    for (int i = 0; i < 2; ++i) {
      int c  = t + i * 256;
      int k  = c >> 3;
      int n0 = (c & 7) * 8;
      v8bf v = *(const v8bf*)(W + (size_t)(k0 + k) * N + bn + n0);
#pragma unroll
      for (int e = 0; e < 8; ++e) sB[(n0 + e) * 64 + k] = v[e];
    }
    if (k0 + 64 < K)  // pull next A slab toward L2/L0 (global_prefetch_b8)
      __builtin_prefetch(A + (size_t)(bm + (t >> 1)) * K + k0 + 64, 0, 3);
    async_fence();
    __syncthreads();

#pragma unroll
    for (int ks = 0; ks < 2; ++ks) {
      v16bf a0 = frag_a(sA, 64, wm * 32,      ks * 32, lane);
      v16bf a1 = frag_a(sA, 64, wm * 32 + 16, ks * 32, lane);
      v16bf b0 = frag_b(sB, 64, wn * 32,      ks * 32, lane);
      v16bf b1 = frag_b(sB, 64, wn * 32 + 16, ks * 32, lane);
      acc[0][0] = WMMA_BF16(a0, b0, acc[0][0]);
      acc[0][1] = WMMA_BF16(a0, b1, acc[0][1]);
      acc[1][0] = WMMA_BF16(a1, b0, acc[1][0]);
      acc[1][1] = WMMA_BF16(a1, b1, acc[1][1]);
    }
    __syncthreads();
  }

  const int hi8 = (lane & 16) ? 8 : 0;
  const int cn  = lane & 15;
#pragma unroll
  for (int i = 0; i < 2; ++i)
#pragma unroll
    for (int j = 0; j < 2; ++j)
#pragma unroll
      for (int r = 0; r < 8; ++r) {
        int m = bm + wm * 32 + i * 16 + r + hi8;
        int n = bn + wn * 32 + j * 16 + cn;
        float v = acc[i][j][r];
        if (do_silu)  // silu via v_rcp_f32, not exact-IEEE divide
          v = v * __builtin_amdgcn_rcpf(1.0f + __expf(-v));
        if (out_f32) {
          out_f32[(size_t)m * N + n] = v;
        } else {
          int h = n >> 6, d = n & 63;
          out_head[((size_t)h * SEQ + m) * 64 + d] = (bf16)v;
        }
      }
}

// ---------------------------------------------------------------------------
// Causal linear attention, one (head, 64-row query block) per workgroup.
//   S = tril(Q K^T); z += rowsum(S)  (== q . cumsum(k));
//   O += S_bf16 @ V; final O * rcp(z+eps).
// 128 threads = 4 waves; wave w owns query rows [16w, 16w+16).
// ---------------------------------------------------------------------------
__global__ __launch_bounds__(128) void ma_attn(
    const bf16* __restrict__ Q, const bf16* __restrict__ Kt,
    const bf16* __restrict__ V,
    float* __restrict__ outf, bf16* __restrict__ outb) {
  __shared__ bf16  Qs[64 * 64];   // [l][d]
  __shared__ bf16  Ks[64 * 64];   // [m][d]  (B-frag for S: col=m, k=d)
  __shared__ bf16  Vt[64 * 64];   // [d][m]  (B-frag for O: col=d, k=m)
  __shared__ bf16  Ss[64 * 64];   // masked scores, bf16
  __shared__ float zsh[64];

  const int t    = threadIdx.x;
  const int lane = t & 31;
  const int wave = t >> 5;
  const int h    = blockIdx.y;
  const int qb   = blockIdx.x;
  const size_t headoff = (size_t)h * SEQ * 64;

#pragma unroll
  for (int i = 0; i < 4; ++i) {              // Q tile: async DMA to LDS
    int c   = t + i * 128;
    int row = c >> 3;
    int off = (c & 7) * 8;
    copy16_to_lds(Q + headoff + (size_t)(qb * 64 + row) * 64 + off,
                  Qs + row * 64 + off);
  }
  if (t < 64) zsh[t] = 0.0f;

  v8f acc[4] = {};
  const int rb  = wave * 16;
  const int hi8 = (lane & 16) ? 8 : 0;
  const int cn  = lane & 15;

  for (int jb = 0; jb <= qb; ++jb) {
    __syncthreads();   // prev-iter Ss/Ks/Vt reads done; zsh init visible
#pragma unroll
    for (int i = 0; i < 4; ++i) {
      int c   = t + i * 128;
      int row = c >> 3;
      int off = (c & 7) * 8;
      copy16_to_lds(Kt + headoff + (size_t)(jb * 64 + row) * 64 + off,
                    Ks + row * 64 + off);
      v8bf v = *(const v8bf*)(V + headoff + (size_t)(jb * 64 + row) * 64 + off);
#pragma unroll
      for (int e = 0; e < 8; ++e) Vt[(off + e) * 64 + row] = v[e];
    }
    async_fence();     // covers Q preamble DMA on first iteration too
    __syncthreads();

    // S strip (16x64) = Q strip @ K^T, contraction over d (64 -> 2 wmma)
#pragma unroll
    for (int nt = 0; nt < 4; ++nt) {
      v8f s = {};
#pragma unroll
      for (int kk = 0; kk < 2; ++kk) {
        v16bf a = frag_a(Qs, 64, rb,      kk * 32, lane);
        v16bf b = frag_b(Ks, 64, nt * 16, kk * 32, lane);
        s = WMMA_BF16(a, b, s);
      }
#pragma unroll
      for (int r = 0; r < 8; ++r) {
        int lrow = rb + r + hi8;
        int lcol = nt * 16 + cn;
        float v = s[r];
        if (jb == qb && lcol > lrow) v = 0.0f;   // causal mask (diag block)
        Ss[lrow * 64 + lcol] = (bf16)v;
      }
    }
    __syncthreads();

    // normalizer: z_l += rowsum(tril(QK^T))_l  == q_l . cumsum(k)_l
    if (t < 64) {
      float az = 0.0f;
#pragma unroll 8
      for (int m = 0; m < 64; ++m) az += (float)Ss[t * 64 + m];
      zsh[t] += az;
    }

    // O strip += S strip @ V, contraction over m (64 -> 2 wmma)
#pragma unroll
    for (int nt = 0; nt < 4; ++nt) {
#pragma unroll
      for (int kk = 0; kk < 2; ++kk) {
        v16bf a = frag_a(Ss, 64, rb,      kk * 32, lane);
        v16bf b = frag_b(Vt, 64, nt * 16, kk * 32, lane);
        acc[nt] = WMMA_BF16(a, b, acc[nt]);
      }
    }
  }
  __syncthreads();

#pragma unroll
  for (int nt = 0; nt < 4; ++nt)
#pragma unroll
    for (int r = 0; r < 8; ++r) {
      int lrow = rb + r + hi8;
      int grow = qb * 64 + lrow;
      int d    = nt * 16 + cn;
      float v  = acc[nt][r] * __builtin_amdgcn_rcpf(zsh[lrow] + 1e-12f);
      if (outf) outf[headoff + (size_t)grow * 64 + d] = v;
      else      outb[(size_t)grow * EMB + h * 64 + d] = (bf16)v;
    }
}

// ---------------------------------------------------------------------------
// Monarch head shuffle: q2[h2][l][h1*4+d2] = y1[h1][l][h2*4+d2]  (fp32->bf16)
// ---------------------------------------------------------------------------
__global__ void ma_shuffle(const float* __restrict__ y1, bf16* __restrict__ q2) {
  int o = blockIdx.x * 256 + threadIdx.x;
  if (o >= SEQ * EMB) return;
  int h2  = o >> 17;                 // / (SEQ*64)
  int rem = o & ((SEQ * 64) - 1);
  int l   = rem >> 6;
  int c   = rem & 63;
  int h1  = c >> 2;
  int d2  = c & 3;
  q2[o] = (bf16)y1[((size_t)h1 * SEQ + l) * 64 + h2 * 4 + d2];
}

// ---------------------------------------------------------------------------
extern "C" void kernel_launch(void* const* d_in, const int* in_sizes, int n_in,
                              void* d_out, int out_size, void* d_ws, size_t ws_size,
                              hipStream_t stream) {
  (void)in_sizes; (void)n_in; (void)out_size; (void)ws_size;
  const float* X  = (const float*)d_in[0];
  const float* Wf[6] = { (const float*)d_in[1], (const float*)d_in[2],
                         (const float*)d_in[3], (const float*)d_in[4],
                         (const float*)d_in[5], (const float*)d_in[6] };

  char* ws = (char*)d_ws;
  size_t off = 0;
  auto alloc = [&](size_t bytes) -> void* {
    void* p = ws + off;
    off = (off + bytes + 255) & ~(size_t)255;
    return p;
  };

  bf16* Xb = (bf16*)alloc((size_t)SEQ * EMB * 2);
  bf16* Wb[6];
  for (int i = 0; i < 6; ++i) Wb[i] = (bf16*)alloc((size_t)EMB * EMB * 2);
  bf16*  qb  = (bf16*)alloc((size_t)SEQ * EMB * 2);   // [H][L][64]
  bf16*  k1b = (bf16*)alloc((size_t)SEQ * EMB * 2);
  bf16*  v1b = (bf16*)alloc((size_t)SEQ * EMB * 2);
  bf16*  k2b = (bf16*)alloc((size_t)SEQ * EMB * 2);
  bf16*  v2b = (bf16*)alloc((size_t)SEQ * EMB * 2);
  float* y1f = (float*)alloc((size_t)SEQ * EMB * 4);  // stage-1 out, fp32
  bf16*  q2b = (bf16*)alloc((size_t)SEQ * EMB * 2);   // shuffled query
  bf16*  y2b = (bf16*)alloc((size_t)SEQ * EMB * 2);   // [L][DIM]

  // 0) converts
  ma_cvt_bf16<<<(SEQ * EMB + 255) / 256, 256, 0, stream>>>(X, Xb, SEQ * EMB);
  for (int i = 0; i < 6; ++i)
    ma_cvt_bf16<<<(EMB * EMB + 255) / 256, 256, 0, stream>>>(Wf[i], Wb[i], EMB * EMB);

  // 1) projections (Wq, Wk1, Wv1, Wk2, Wv2)
  dim3 ggrid(EMB / 64, SEQ / 128);
  ma_gemm_bf16<<<ggrid, 256, 0, stream>>>(Xb, Wb[0], SEQ, EMB, EMB, 1, qb,  nullptr);
  ma_gemm_bf16<<<ggrid, 256, 0, stream>>>(Xb, Wb[1], SEQ, EMB, EMB, 1, k1b, nullptr);
  ma_gemm_bf16<<<ggrid, 256, 0, stream>>>(Xb, Wb[2], SEQ, EMB, EMB, 0, v1b, nullptr);
  ma_gemm_bf16<<<ggrid, 256, 0, stream>>>(Xb, Wb[3], SEQ, EMB, EMB, 1, k2b, nullptr);
  ma_gemm_bf16<<<ggrid, 256, 0, stream>>>(Xb, Wb[4], SEQ, EMB, EMB, 0, v2b, nullptr);

  // 2) stage-1 causal linear attention -> y1 (fp32 head layout)
  dim3 agrid(SEQ / 64, NHEAD);
  ma_attn<<<agrid, 128, 0, stream>>>(qb, k1b, v1b, y1f, nullptr);

  // 3) Monarch head shuffle -> q2 (bf16)
  ma_shuffle<<<(SEQ * EMB + 255) / 256, 256, 0, stream>>>(y1f, q2b);

  // 4) stage-2 causal linear attention -> y2 (bf16 [L][DIM])
  ma_attn<<<agrid, 128, 0, stream>>>(q2b, k2b, v2b, nullptr, y2b);

  // 5) output projection -> fp32 d_out
  ma_gemm_bf16<<<ggrid, 256, 0, stream>>>(y2b, Wb[5], SEQ, EMB, EMB, 0,
                                          nullptr, (float*)d_out);
}
